// VariationalLambdaGW_56959856280145
// MI455X (gfx1250) — matrix operand order repaired and checked by reading, not compiled, not run on hardware
//
#include <hip/hip_runtime.h>
#include <hip/hip_bf16.h>
#include <math.h>

// ---------------------------------------------------------------------------
// CDNA5 / gfx1250 implementation of VariationalLambdaGW forward.
//
// Roofline: conv stacks dominate (~233 GF of the ~280 GF total) at ~480 FLOP/B
// arithmetic intensity -> matrix-core bound, so all matmul-shaped work runs on
// v_wmma_f32_16x16x32_f16 (wave32 WMMA, f32 accumulate). Each wave computes a
// 16x32 tile (2 WMMAs / k-step, A-fragment reused). A-fragments are built
// directly from global memory when A is k-contiguous (the f16 A layout is two
// contiguous 8-element runs per lane). B panels are staged to LDS by the
// Tensor Data Mover (tensor_load_to_lds + s_wait_tensorcnt) when available,
// with a branchless clamp/select fallback so loads batch behind one wait.
// ---------------------------------------------------------------------------

typedef __attribute__((ext_vector_type(16))) _Float16 v16h;
typedef __attribute__((ext_vector_type(8)))  float    v8f;

#define B_SZ   16
#define T1D    1024
#define T2D    1024
#define NCH    384
#define LDIMD  16

// ---------------- TDM support (compile-guarded) ----------------------------
#if __has_builtin(__builtin_amdgcn_tensor_load_to_lds) && __has_builtin(__builtin_amdgcn_s_wait_tensorcnt)
#define HAVE_TDM 1
typedef __attribute__((ext_vector_type(4))) unsigned int u32x4;
typedef __attribute__((ext_vector_type(8))) int i32x8;
typedef __attribute__((ext_vector_type(4))) int i32x4;

// Load a rows x cols f32 tile (row stride = strideElems) from global into LDS.
// D# layout per CDNA5 ISA 8.3/8.4 (2D tensor, groups 2/3 zero).
__device__ __forceinline__ void tdm_load_tile_f32(const float* gsrc, void* ldst,
                                                  unsigned rows, unsigned cols,
                                                  unsigned strideElems) {
  unsigned long long ga = (unsigned long long)(size_t)gsrc;
  unsigned lds = (unsigned)(size_t)ldst;          // low 32 bits = LDS byte offset
  u32x4 g0;
  g0[0] = 1u;                                     // count=1, user descriptor
  g0[1] = lds;                                    // lds_addr  (bits 63:32)
  g0[2] = (unsigned)(ga & 0xFFFFFFFFu);           // global_addr[31:0]
  g0[3] = (unsigned)((ga >> 32) & 0x01FFFFFFu) | (2u << 30); // addr[56:32] | type=2
  i32x8 g1;
  g1[0] = (int)(2u << 16);                        // data_size = 4B
  g1[1] = (int)((cols & 0xFFFFu) << 16);          // tensor_dim0[15:0]
  g1[2] = (int)((rows & 0xFFFFu) << 16);          // dim0 hi=0 | tensor_dim1[15:0]
  g1[3] = (int)((cols & 0xFFFFu) << 16);          // dim1 hi=0 | tile_dim0
  g1[4] = (int)(rows & 0xFFFFu);                  // tile_dim1 | tile_dim2=0
  g1[5] = (int)strideElems;                       // tensor_dim0_stride[31:0]
  g1[6] = 0;                                      // stride hi | dim1_stride lo
  g1[7] = 0;
  i32x4 z4 = {0, 0, 0, 0};
#if defined(__clang_major__) && (__clang_major__ >= 23)
  i32x8 z8 = {0, 0, 0, 0, 0, 0, 0, 0};
  __builtin_amdgcn_tensor_load_to_lds(g0, g1, z4, z4, z8, 0);
#else
  __builtin_amdgcn_tensor_load_to_lds(g0, g1, z4, z4, 0);
#endif
}
#else
#define HAVE_TDM 0
#endif

// ---------------- WMMA fragment helpers (ISA 7.12.2 wave32 layouts) --------
// A 16x32 f16: lane l holds row m=l&15, k-runs {khi..khi+7, khi+16..khi+23}
__device__ __forceinline__ v16h frag_a_from_lds(const _Float16* As, int lane) {
  v16h a;
  int m = lane & 15;
  int khi = (lane & 16) ? 8 : 0;
#pragma unroll
  for (int h = 0; h < 16; ++h) {
    int k = khi + (h & 7) + ((h >> 3) << 4);
    a[h] = As[m * 32 + k];
  }
  return a;
}
// B 32x16 f16: lane l holds col n=l&15, k-runs {kb..kb+7, kb+16..kb+23} (kb = 0 or 8); tile f32 in LDS
__device__ __forceinline__ v16h frag_b_f32lds(const float* Bsf, int ncol, int lane) {
  v16h b;
  int kb = (lane & 16) ? 8 : 0;
#pragma unroll
  for (int h = 0; h < 16; ++h) b[h] = (_Float16)Bsf[(kb + (h & 7) + ((h >> 3) << 4)) * 32 + ncol];
  return b;
}

// ---------------- Generic strided / batched WMMA GEMM (16x32 per wave) -----
// C[z, m, n] (ldC = N) = A[z, m*sAm + k*sAk] * Bm[z, k*sBk + n*sBn] (+bias)(+C)
__global__ void gemm_wmma_kernel(const float* __restrict__ A, const float* __restrict__ Bm,
                                 const float* __restrict__ bias, float* __restrict__ C,
                                 int M, int N, int K,
                                 int sAm, int sAk, int sBk, int sBn,
                                 long long Abs, long long Bbs, long long Cbs,
                                 int accum) {
  __shared__ _Float16 As[16 * 32];
  __shared__ float Bsf[32 * 32];
  const int lane = threadIdx.x;
  const int z = blockIdx.z;
  const float* Az = A + (size_t)z * Abs;
  const float* Bz = Bm + (size_t)z * Bbs;
  float* Cz = C + (size_t)z * Cbs;
  const int m0 = blockIdx.x << 4, n0 = blockIdx.y << 5;
  const bool a_fast = (sAk == 1);
  const int m = lane & 15;
  const int khi = (lane & 16) ? 8 : 0;
  const int mrow = m0 + m;
  const bool mok = mrow < M;
  const float* arow = Az + (size_t)(mok ? mrow : (M - 1)) * sAm;
#if HAVE_TDM
  const bool tdm_ok = (sBn == 1) && (n0 + 32 <= N);
#endif
  v8f acc0 = {}, acc1 = {};
  for (int k0 = 0; k0 < K; k0 += 32) {
    // ---- stage B panel (32k x 32n, f32) into LDS ----
#if HAVE_TDM
    if (tdm_ok && (k0 + 32 <= K)) {
      tdm_load_tile_f32(Bz + (size_t)k0 * sBk + n0, Bsf, 32u, 32u, (unsigned)sBk);
      __builtin_amdgcn_s_wait_tensorcnt(0);
    } else
#endif
    {
      int k = k0 + lane;
      float* dst = Bsf + lane * 32;
      if (k < K && n0 + 32 <= N) {
        const float* brow = Bz + (size_t)k * sBk + (size_t)n0 * sBn;
        if (sBn == 1) {
#pragma unroll
          for (int i = 0; i < 32; ++i) dst[i] = brow[i];
        } else {
#pragma unroll
          for (int i = 0; i < 32; ++i) dst[i] = brow[(size_t)i * sBn];
        }
      } else {
        int kc = k < K ? k : K - 1;
        const float* brow = Bz + (size_t)kc * sBk;
#pragma unroll
        for (int i = 0; i < 32; ++i) {
          int n = n0 + i;
          int nc = n < N ? n : N - 1;
          float v = brow[(size_t)nc * sBn];
          dst[i] = (k < K && n < N) ? v : 0.f;
        }
      }
    }
    // ---- A fragment ----
    v16h af;
    if (a_fast) {
      if (mok && (k0 + 32 <= K)) {
        const float* p = arow + k0 + khi;
        if (k0 + 64 <= K) __builtin_prefetch(p + 32, 0, 1);
#pragma unroll
        for (int h = 0; h < 8; ++h) af[h] = (_Float16)p[h];
#pragma unroll
        for (int h = 0; h < 8; ++h) af[8 + h] = (_Float16)p[16 + h];
      } else {
#pragma unroll
        for (int h = 0; h < 16; ++h) {
          int k = k0 + khi + (h & 7) + ((h >> 3) << 4);
          float v = arow[k < K ? k : K - 1];
          af[h] = (_Float16)((mok && k < K) ? v : 0.f);
        }
      }
      __syncthreads();                 // B staging visible
    } else {
      { // branchless LDS staging of A (clamped loads, batched issue)
        int ks = (lane >> 4) << 4;
#pragma unroll
        for (int i = 0; i < 16; ++i) {
          int k = k0 + ks + i;
          float v = arow[(size_t)(k < K ? k : K - 1) * sAk];
          As[m * 32 + ks + i] = (_Float16)((mok && k < K) ? v : 0.f);
        }
      }
      __syncthreads();
      af = frag_a_from_lds(As, lane);
    }
    // ---- two WMMAs, reusing the A fragment ----
    v16h bf0 = frag_b_f32lds(Bsf, lane & 15, lane);
    v16h bf1 = frag_b_f32lds(Bsf, (lane & 15) + 16, lane);
    acc0 = __builtin_amdgcn_wmma_f32_16x16x32_f16(false, af, false, bf0, (short)0, acc0, false, false);
    acc1 = __builtin_amdgcn_wmma_f32_16x16x32_f16(false, af, false, bf1, (short)0, acc1, false, false);
    __syncthreads();
  }
  // ---- store (C/D layout: VGPR r -> M = r + 8*(lane>=16), N = lane&15) ----
  const int n = lane & 15;
  const int mb = (lane >> 4) << 3;
#pragma unroll
  for (int half = 0; half < 2; ++half) {
    int nn = n0 + n + (half << 4);
    if (nn >= N) continue;
    float bv = bias ? bias[nn] : 0.f;
    const v8f& a = half ? acc1 : acc0;
#pragma unroll
    for (int r = 0; r < 8; ++r) {
      int mm = m0 + mb + r;
      if (mm < M) {
        size_t idx = (size_t)mm * N + nn;
        float o = a[r] + bv;
        if (accum) o += Cz[idx];
        Cz[idx] = o;
      }
    }
  }
}

// ---------------- Implicit-im2col Conv1d (K=5, same pad) via WMMA ----------
// X:(B,T,Cin) row-major, W:(Cout,Cin,5), Yp:(B,T,Cout). k = dk*Cin + cin.
__global__ void conv_wmma_kernel(const float* __restrict__ X, const float* __restrict__ W,
                                 const float* __restrict__ bias, float* __restrict__ Yp,
                                 int T, int Cin, int Cout) {
  __shared__ _Float16 As[16 * 32];
  __shared__ float Bsf[32 * 32];
  const int lane = threadIdx.x;
  const int tilesPerB = T >> 4;
  const int tt = blockIdx.x;
  const int b = tt / tilesPerB;
  const int t0 = (tt - b * tilesPerB) << 4;
  const int n0 = blockIdx.y << 5;
  const int KK = Cin * 5;
  v8f acc0 = {}, acc1 = {};
  for (int k0 = 0; k0 < KK; k0 += 32) {
    { // A: branchless im2col gather (clamped loads, select zero)
      int m = lane & 15, ks = (lane >> 4) << 4;
      int t = t0 + m;
#pragma unroll
      for (int i = 0; i < 16; ++i) {
        int k = k0 + ks + i;
        int kc = k < KK ? k : KK - 1;
        int dk = kc / Cin, cin = kc - dk * Cin;
        int tp = t + dk - 2;
        int tpc = tp < 0 ? 0 : (tp >= T ? T - 1 : tp);
        float v = X[((size_t)b * T + tpc) * Cin + cin];
        As[m * 32 + ks + i] = (_Float16)((k < KK && tp >= 0 && tp < T) ? v : 0.f);
      }
    }
    { // B: weights W[n, cin, dk] (n-stride = Cin*5, not TDM-friendly)
      int k = k0 + lane;
      int kc = k < KK ? k : KK - 1;
      int dk = kc / Cin, cin = kc - dk * Cin;
      const float* wbase = W + (size_t)cin * 5 + dk;
      float* dst = Bsf + lane * 32;
#pragma unroll
      for (int i = 0; i < 32; ++i) {
        float v = wbase[(size_t)(n0 + i) * Cin * 5];
        dst[i] = (k < KK) ? v : 0.f;
      }
    }
    __syncthreads();
    v16h af = frag_a_from_lds(As, lane);
    v16h bf0 = frag_b_f32lds(Bsf, lane & 15, lane);
    v16h bf1 = frag_b_f32lds(Bsf, (lane & 15) + 16, lane);
    acc0 = __builtin_amdgcn_wmma_f32_16x16x32_f16(false, af, false, bf0, (short)0, acc0, false, false);
    acc1 = __builtin_amdgcn_wmma_f32_16x16x32_f16(false, af, false, bf1, (short)0, acc1, false, false);
    __syncthreads();
  }
  const int n = lane & 15;
  const int mb = (lane >> 4) << 3;
#pragma unroll
  for (int half = 0; half < 2; ++half) {
    int nn = n0 + n + (half << 4);
    if (nn >= Cout) continue;
    float bv = bias ? bias[nn] : 0.f;
    const v8f& a = half ? acc1 : acc0;
#pragma unroll
    for (int r = 0; r < 8; ++r) {
      int t = t0 + mb + r;
      Yp[((size_t)b * T + t) * Cout + nn] = a[r] + bv;
    }
  }
}

// ---------------- (optional ReLU) + LayerNorm over last dim ----------------
__global__ void relu_ln_kernel(const float* __restrict__ X, const float* __restrict__ g,
                               const float* __restrict__ beta, float* __restrict__ Y,
                               int C, int do_relu, int rpb_in, int rpb_out, int row_off) {
  __shared__ float s_sum[128];
  __shared__ float s_sq[128];
  const int r = blockIdx.x;
  const int b = r / rpb_in, t = r - b * rpb_in;
  const float* x = X + (size_t)r * C;
  float* y = Y + ((size_t)b * rpb_out + row_off + t) * C;
  const int tid = threadIdx.x;
  float vals[4];
  const int nit = (C + 127) >> 7;
  float lsum = 0.f, lsq = 0.f;
  for (int i = 0; i < nit; ++i) {
    int c = (i << 7) + tid;
    float v = 0.f;
    if (c < C) { v = x[c]; if (do_relu) v = fmaxf(v, 0.f); }
    vals[i] = v; lsum += v; lsq += v * v;
  }
  s_sum[tid] = lsum; s_sq[tid] = lsq;
  __syncthreads();
  for (int s = 64; s > 0; s >>= 1) {
    if (tid < s) { s_sum[tid] += s_sum[tid + s]; s_sq[tid] += s_sq[tid + s]; }
    __syncthreads();
  }
  const float mean = s_sum[0] / (float)C;
  const float var = s_sq[0] / (float)C - mean * mean;
  const float rstd = rsqrtf(var + 1e-5f);
  for (int i = 0; i < nit; ++i) {
    int c = (i << 7) + tid;
    if (c < C) y[c] = (vals[i] - mean) * rstd * g[c] + beta[c];
  }
}

// ---------------- softmax over time (axis -2), in place, with mask ---------
__global__ void softmax_time_kernel(float* __restrict__ Ks, int T, int split,
                                    const int* __restrict__ ilens, const int* __restrict__ olens) {
  __shared__ float red[256];
  const int b = blockIdx.x >> 4, l = blockIdx.x & 15;
  float* col = Ks + (size_t)b * T * 16 + l;
  const int len1 = ilens[b], len2 = olens[b];
  const int tid = threadIdx.x;
  float mx = -3.402823466e+38f;
  for (int t = tid; t < T; t += 256) {
    bool masked = (t < split) ? (t >= len1) : ((t - split) >= len2);
    float v = masked ? -3.402823466e+38f : col[(size_t)t * 16];
    mx = fmaxf(mx, v);
  }
  red[tid] = mx; __syncthreads();
  for (int s = 128; s > 0; s >>= 1) { if (tid < s) red[tid] = fmaxf(red[tid], red[tid + s]); __syncthreads(); }
  mx = red[0]; __syncthreads();
  float sum = 0.f;
  for (int t = tid; t < T; t += 256) {
    bool masked = (t < split) ? (t >= len1) : ((t - split) >= len2);
    float v = masked ? -3.402823466e+38f : col[(size_t)t * 16];
    sum += expf(v - mx);
  }
  red[tid] = sum; __syncthreads();
  for (int s = 128; s > 0; s >>= 1) { if (tid < s) red[tid] += red[tid + s]; __syncthreads(); }
  sum = red[0];
  const float inv = 1.f / sum;
  for (int t = tid; t < T; t += 256) {
    bool masked = (t < split) ? (t >= len1) : ((t - split) >= len2);
    float v = masked ? -3.402823466e+38f : col[(size_t)t * 16];
    col[(size_t)t * 16] = expf(v - mx) * inv;
  }
}

// ---------------- v += pos_enc(t,384); masked -> 0 --------------------------
__global__ void v_pe_mask_kernel(float* __restrict__ V, int T, int split,
                                 const int* __restrict__ ilens, const int* __restrict__ olens, int n) {
  int idx = blockIdx.x * blockDim.x + threadIdx.x;
  if (idx >= n) return;
  const int c = idx % NCH;
  const int bt = idx / NCH;
  const int t = bt % T, b = bt / T;
  bool masked = (t < split) ? (t >= ilens[b]) : ((t - split) >= olens[b]);
  if (masked) { V[idx] = 0.f; return; }
  const float tt = (float)((t < split) ? t : t - split);
  float pe;
  if (c < 192) {
    float dv = expf((float)(2 * c) * (-7.8240460108562919f / 384.f));
    pe = sinf(tt * dv);
  } else {
    int d = c - 192;
    float dv = expf((float)(2 * d) * (-7.8240460108562919f / 384.f));
    pe = cosf(tt * dv);
  }
  V[idx] += pe * 0.072168783648703220f; // 1/sqrt(192)
}

// ---------------- s = pos_enc(t,80) @ We + be ------------------------------
__global__ void posenc_emb_kernel(const float* __restrict__ tpos, const float* __restrict__ We,
                                  const float* __restrict__ be, float* __restrict__ S, int n) {
  int idx = blockIdx.x * blockDim.x + threadIdx.x;
  if (idx >= n) return;
  const int j = idx & 15;
  const int row = idx >> 4;
  const float tt = tpos[row];
  float acc = be[j];
#pragma unroll 4
  for (int d = 0; d < 40; ++d) {
    float dv = expf((float)(2 * d) * (-7.8240460108562919f / 80.f));
    float a = tt * dv;
    acc += (sinf(a) * We[d * 16 + j] + cosf(a) * We[(40 + d) * 16 + j]) * 0.15811388300841897f;
  }
  S[idx] = acc;
}

// ---------------- reparameterized sample l = mu + eps*exp(logs) ------------
__global__ void l_sample_kernel(const float* __restrict__ lq, const float* __restrict__ eps,
                                float* __restrict__ L, int n) {
  int idx = blockIdx.x * blockDim.x + threadIdx.x;
  if (idx >= n) return;
  L[idx] = lq[2 * idx] + eps[idx] * expf(lq[2 * idx + 1]);
}

// ---------------- RK4 plumbing ---------------------------------------------
__global__ void finit_kernel(float* __restrict__ f, int n, int T) {
  int idx = blockIdx.x * blockDim.x + threadIdx.x;
  if (idx < n) f[idx] = (float)(idx % T);
}
__global__ void axpby_kernel(const float* __restrict__ f, const float* __restrict__ k,
                             float s, float* __restrict__ o, int n) {
  int idx = blockIdx.x * blockDim.x + threadIdx.x;
  if (idx < n) o[idx] = f[idx] + s * k[idx];
}
__global__ void fcombine_kernel(float* __restrict__ f, const float* __restrict__ k1,
                                const float* __restrict__ k2, const float* __restrict__ k3,
                                const float* __restrict__ k4, int n) {
  int idx = blockIdx.x * blockDim.x + threadIdx.x;
  if (idx < n) f[idx] += (2.f / 3.f) * (k1[idx] + 2.f * k2[idx] + 2.f * k3[idx] + k4[idx]);
}
__global__ void fscale_kernel(float* __restrict__ f, float* __restrict__ fout,
                              const int* __restrict__ ilens, const int* __restrict__ olens,
                              int T, int n) {
  int idx = blockIdx.x * blockDim.x + threadIdx.x;
  if (idx >= n) return;
  int b = idx / T;
  float v = f[idx] * ((float)ilens[b] / (float)olens[b]);
  f[idx] = v; fout[idx] = v;
}

// ---------------- Catmull-Rom cubic interpolation --------------------------
__global__ void cubic_kernel(const float* __restrict__ x, const float* __restrict__ f,
                             float* __restrict__ y, int n) {
  int idx = blockIdx.x * blockDim.x + threadIdx.x;
  if (idx >= n) return;
  const int c = idx % 80;
  const int bt = idx / 80;
  const int t2 = bt % T2D, b = bt / T2D;
  const float ff = f[(size_t)b * T2D + t2];
  const float fl = floorf(ff);
  const int i0 = (int)fl;
  const float u = ff - fl;
  float xx[4];
#pragma unroll
  for (int o = 0; o < 4; ++o) {
    int i = i0 + o - 1;
    i = i < 0 ? 0 : (i > T1D - 1 ? T1D - 1 : i);
    xx[o] = x[((size_t)b * T1D + i) * 80 + c];
  }
  const float u2 = u * u, u3 = u2 * u;
  y[idx] = 0.5f * (2.f * xx[1] + (xx[2] - xx[0]) * u +
                   (2.f * xx[0] - 5.f * xx[1] + 4.f * xx[2] - xx[3]) * u2 +
                   (-xx[0] + 3.f * xx[1] - 3.f * xx[2] + xx[3]) * u3);
}

// ===========================================================================
extern "C" void kernel_launch(void* const* d_in, const int* in_sizes, int n_in,
                              void* d_out, int out_size, void* d_ws, size_t ws_size,
                              hipStream_t stream) {
  auto P = [&](int i) -> const float* { return (const float*)d_in[i]; };
  const float* x = P(0);                 // (16,1024,80)
  const float* yv = P(1);                // (16,1024,80)
  const int* ilens = (const int*)d_in[2];
  const int* olens = (const int*)d_in[3];
  const float* eps = P(4);               // (16,16,16,4,4)
  // params flattened in sorted-dict / pytree order (see round-0 analysis)
  const int CONV_PRIOR = 5, CONV_PX = 21, CONV_PY = 37;
  const int WE = 53, BE = 54;
  const int LAMP = 55, LAMQ = 59;
  const int LINS = 63;
  const int LP = 87, LQ = 105;
  const int NORMS_G = 123, NORMS_B = 124;

  float* ws = (float*)d_ws;
  float* out = (float*)d_out;

  size_t off = 0;
  auto take = [&](size_t nel) { size_t r = off; off += nel; return r; };
  const size_t convA_o = take((size_t)B_SZ * T1D * NCH);
  const size_t convB_o = take((size_t)B_SZ * T1D * NCH);
  const size_t p_o     = take((size_t)B_SZ * T1D * NCH);
  const size_t q_o     = take((size_t)B_SZ * 2 * T1D * NCH);
  const size_t kq_o    = take((size_t)B_SZ * 2 * T1D * 16);
  const size_t poolp_o = take((size_t)B_SZ * 16 * NCH);
  const size_t poolq_o = take((size_t)B_SZ * 16 * NCH);
  const size_t lsa_o   = take((size_t)B_SZ * 16 * NCH);
  const size_t lsb_o   = take((size_t)B_SZ * 16 * NCH);
  const size_t l_o     = take((size_t)B_SZ * 16 * 16 * 16);
  const size_t s_o     = take((size_t)B_SZ * T2D * 16);
  const size_t sn_o    = take((size_t)B_SZ * T2D * 16);
  const size_t ks_o    = take((size_t)4 * B_SZ * T2D);
  const size_t f_o     = take((size_t)B_SZ * T2D);
  const size_t ft_o    = take((size_t)B_SZ * T2D);
  (void)ws_size; (void)convB_o; // ~131 MB of f32 scratch required

  float* p_buf = ws + p_o;
  float* q_buf = ws + q_o;
  float* kq    = ws + kq_o;
  float* poolp = ws + poolp_o;
  float* poolq = ws + poolq_o;
  float* ls_a  = ws + lsa_o;
  float* ls_b  = ws + lsb_o;
  float* l_buf = ws + l_o;
  float* s_buf = ws + s_o;
  float* sn_buf = ws + sn_o;
  float* ksig  = ws + ks_o;
  float* f_buf = ws + f_o;
  float* ftmp  = ws + ft_o;
  float* convA = ws + convA_o;
  float* convB = ws + convB_o;
  float* v_buf = ws + convA_o;  // reuse convA+convB (conv stacks finished)
  float* h_buf = ws + q_o;      // reuse q (lam finished by then)

  float* out_y  = out;                                      // (16,1024,80)
  float* out_f  = out + (size_t)B_SZ * T2D * 80;            // (16,1024)
  float* out_lp = out_f + (size_t)B_SZ * T2D;               // (16,4096,2)
  float* out_lq = out_lp + (size_t)B_SZ * 4096 * 2;         // (16,4096,2)

  const int NF = B_SZ * T2D;

  auto gemm = [&](const float* A, const float* Bm, const float* bias, float* C,
                  int M, int N, int K, int sAm, int sAk, int sBk, int sBn,
                  long long Abs, long long Bbs, long long Cbs, int nz, int accum) {
    dim3 grid((M + 15) / 16, (N + 31) / 32, nz);
    gemm_wmma_kernel<<<grid, 32, 0, stream>>>(A, Bm, bias, C, M, N, K,
                                              sAm, sAk, sBk, sBn, Abs, Bbs, Cbs, accum);
  };

  // ---- conv stacks (implicit im2col WMMA + relu/LN over channels) ----------
  auto conv_stack = [&](const float* xin, int pi, float* finalOut, int rpb_out, int row_off) {
    const float* cur = xin;
    int Cin = 80;
    for (int L = 0; L < 4; ++L) {
      dim3 grid(B_SZ * T1D / 16, NCH / 32);
      conv_wmma_kernel<<<grid, 32, 0, stream>>>(cur, P(pi + 4 * L), P(pi + 4 * L + 1),
                                                convA, T1D, Cin, NCH);
      float* dst = (L == 3) ? finalOut : convB;
      int rpo = (L == 3) ? rpb_out : T1D;
      int ro = (L == 3) ? row_off : 0;
      relu_ln_kernel<<<B_SZ * T1D, 128, 0, stream>>>(convA, P(pi + 4 * L + 2), P(pi + 4 * L + 3),
                                                     dst, NCH, 1, T1D, rpo, ro);
      cur = convB;
      Cin = NCH;
    }
  };
  conv_stack(x, CONV_PRIOR, p_buf, T1D, 0);
  conv_stack(x, CONV_PX, q_buf, 2 * T1D, 0);
  conv_stack(yv, CONV_PY, q_buf, 2 * T1D, T1D);

  // ---- lambda attention pooling -------------------------------------------
  auto lam = [&](const float* xb, int T, int pbase, float* pool) {
    gemm(xb, P(pbase + 0), P(pbase + 2), kq, B_SZ * T, 16, NCH, NCH, 1, 16, 1, 0, 0, 0, 1, 0);
    softmax_time_kernel<<<B_SZ * 16, 256, 0, stream>>>(kq, T, T1D, ilens, olens);
    gemm(xb, P(pbase + 1), P(pbase + 3), v_buf, B_SZ * T, NCH, NCH, NCH, 1, NCH, 1, 0, 0, 0, 1, 0);
    int nv = B_SZ * T * NCH;
    v_pe_mask_kernel<<<(nv + 255) / 256, 256, 0, stream>>>(v_buf, T, T1D, ilens, olens, nv);
    // pool[b,l,c] = sum_t scores[b,t,l] * v[b,t,c]  (A = scores^T via strides)
    gemm(kq, v_buf, nullptr, pool, 16, NCH, T, 1, 16, NCH, 1,
         (long long)T * 16, (long long)T * NCH, (long long)16 * NCH, B_SZ, 0);
  };
  lam(p_buf, T1D, LAMP, poolp);
  lam(q_buf, 2 * T1D, LAMQ, poolq);

  // ---- lin stacks -> lp / lq written straight into d_out ------------------
  auto lin_stack = [&](const float* pool, int base, float* dst) {
    const float* cur = pool;
    for (int L = 0; L < 4; ++L) {
      gemm(cur, P(base + 4 * L), P(base + 4 * L + 1), ls_a, B_SZ * 16, NCH, NCH,
           NCH, 1, NCH, 1, 0, 0, 0, 1, 0);
      relu_ln_kernel<<<B_SZ * 16, 128, 0, stream>>>(ls_a, P(base + 4 * L + 2), P(base + 4 * L + 3),
                                                    ls_b, NCH, 1, B_SZ * 16, B_SZ * 16, 0);
      cur = ls_b;
    }
    gemm(cur, P(base + 16), P(base + 17), dst, B_SZ * 16, 512, NCH, NCH, 1, 512, 1, 0, 0, 0, 1, 0);
  };
  lin_stack(poolp, LP, out_lp);
  lin_stack(poolq, LQ, out_lq);

  // ---- l = mu + eps*exp(logs) ---------------------------------------------
  {
    int n = B_SZ * 4096;
    l_sample_kernel<<<(n + 255) / 256, 256, 0, stream>>>(out_lq, eps, l_buf, n);
  }

  // ---- forward_function: NCOMP x RK4 of gw_signal -------------------------
  finit_kernel<<<(NF + 255) / 256, 256, 0, stream>>>(f_buf, NF, T2D);

  auto gw_signal = [&](const float* tpos, int comp, float* kout) {
    int n = B_SZ * T2D * 16;
    posenc_emb_kernel<<<(n + 127) / 128, 128, 0, stream>>>(tpos, P(WE), P(BE), s_buf, n);
    for (int L = 0; L < 4; ++L) {
      relu_ln_kernel<<<B_SZ * T2D, 128, 0, stream>>>(s_buf, P(NORMS_G), P(NORMS_B), sn_buf,
                                                     16, 0, T2D, T2D, 0);
      // s = res + sn @ l[b,:,:,L,comp]   (batched 16x16, accumulate into s)
      gemm(sn_buf, l_buf + (L * 4 + comp), nullptr, s_buf, T2D, 16, 16,
           16, 1, 256, 16, (long long)T2D * 16, 4096, (long long)T2D * 16, B_SZ, 1);
      gemm(s_buf, P(LINS + 6 * L), P(LINS + 6 * L + 1), h_buf, B_SZ * T2D, NCH, 16,
           16, 1, NCH, 1, 0, 0, 0, 1, 0);
      relu_ln_kernel<<<B_SZ * T2D, 128, 0, stream>>>(h_buf, P(LINS + 6 * L + 2), P(LINS + 6 * L + 3),
                                                     h_buf, NCH, 1, B_SZ * T2D, B_SZ * T2D, 0);
      int dO = (L == 3) ? 1 : 16;
      gemm(h_buf, P(LINS + 6 * L + 4), P(LINS + 6 * L + 5), (L == 3) ? kout : s_buf,
           B_SZ * T2D, dO, NCH, NCH, 1, dO, 1, 0, 0, 0, 1, 0);
    }
    // raw k; the x4 ( /NCOMP * 16 ) factor is folded into RK4 combine scalars.
  };

  const int nfb = (NF + 255) / 256;
  for (int comp = 0; comp < 4; ++comp) {
    float* k1 = ksig + 0 * NF;
    float* k2 = ksig + 1 * NF;
    float* k3 = ksig + 2 * NF;
    float* k4 = ksig + 3 * NF;
    gw_signal(f_buf, comp, k1);
    axpby_kernel<<<nfb, 256, 0, stream>>>(f_buf, k1, 2.f, ftmp, NF); // f + (4k1)/2
    gw_signal(ftmp, comp, k2);
    axpby_kernel<<<nfb, 256, 0, stream>>>(f_buf, k2, 2.f, ftmp, NF); // f + (4k2)/2
    gw_signal(ftmp, comp, k3);
    axpby_kernel<<<nfb, 256, 0, stream>>>(f_buf, k3, 4.f, ftmp, NF); // f + 4k3
    gw_signal(ftmp, comp, k4);
    fcombine_kernel<<<nfb, 256, 0, stream>>>(f_buf, k1, k2, k3, k4, NF);
  }

  // ---- final scale + outputs ----------------------------------------------
  fscale_kernel<<<nfb, 256, 0, stream>>>(f_buf, out_f, ilens, olens, T2D, NF);
  {
    int n = B_SZ * T2D * 80;
    cubic_kernel<<<(n + 255) / 256, 256, 0, stream>>>(x, f_buf, out_y, n);
  }
}
